// GGCN_89026082111503
// MI455X (gfx1250) — matrix operand order repaired and checked by reading, not compiled, not run on hardware
//
#include <hip/hip_runtime.h>
#include <hip/hip_bf16.h>
#include <math.h>

#define N_NODES 100000
#define N_EDGES 3200000
#define NFEAT   512
#define NHID    256
#define NCLASS  16
#define NLAYERS 8

typedef __attribute__((ext_vector_type(16))) __bf16 v16bf;
typedef __attribute__((ext_vector_type(8)))  __bf16 v8bf;
typedef __attribute__((ext_vector_type(8)))  float  v8f;
typedef int v4i_vs __attribute__((vector_size(16)));   // matches builtin's int __vector(4)

// ---------------------------------------------------------------------------
// 16-element bf16 fragment load as two 16B chunks (works for global and LDS;
// LDS reads lower to ds_load_b128).
// ---------------------------------------------------------------------------
__device__ __forceinline__ v16bf load16(const __bf16* p) {
    v8bf lo = *(const v8bf*)(p);
    v8bf hi = *(const v8bf*)(p + 16);
    v16bf a;
#pragma unroll
    for (int i = 0; i < 8; ++i) { a[i] = lo[i]; a[i + 8] = hi[i]; }
    return a;
}

// A-fragment: 16x32 bf16 tile per ISA 7.12.2 layout.
// lane m = lane&15, h = lane>>4:
//   elems 0..7  = A[m][k0 + 8h      .. +8)
//   elems 8..15 = A[m][k0 + 16 + 8h .. +8)
__device__ __forceinline__ v16bf load_a_frag(const __bf16* Arow, int k0, int half) {
    v8bf lo = *(const v8bf*)(Arow + k0 + 8 * half);
    v8bf hi = *(const v8bf*)(Arow + k0 + 16 + 8 * half);
    v16bf a;
#pragma unroll
    for (int i = 0; i < 8; ++i) { a[i] = lo[i]; a[i + 8] = hi[i]; }
    return a;
}

// ---------------------------------------------------------------------------
// Conversions / transposes (one-time, tiny vs HBM budget)
// ---------------------------------------------------------------------------
__global__ __launch_bounds__(256) void cvt_f32_bf16_kernel(const float* __restrict__ in,
                                                           __bf16* __restrict__ out, int n) {
    int i = blockIdx.x * 256 + threadIdx.x;
    if (i < n) out[i] = (__bf16)in[i];
}

// w  : [K, N] f32 row-major  ->  wt : [N, K] bf16 row-major
__global__ __launch_bounds__(256) void transpose_bf16_kernel(const float* __restrict__ w,
                                                             __bf16* __restrict__ wt,
                                                             int K, int N) {
    int i = blockIdx.x * 256 + threadIdx.x;
    if (i < K * N) {
        int k = i / N, n = i % N;
        wt[(size_t)n * K + k] = (__bf16)w[i];
    }
}

// conv_w : [L, K, N] -> cwt : [L, N, K] bf16
__global__ __launch_bounds__(256) void transpose_cw_kernel(const float* __restrict__ w,
                                                           __bf16* __restrict__ wt) {
    int i = blockIdx.x * 256 + threadIdx.x;
    int l = i >> 16;             // / (256*256)
    int r = (i >> 8) & 255;      // k
    int c = i & 255;             // n
    wt[((size_t)l << 16) + ((size_t)c << 8) + r] = (__bf16)w[i];
}

// ---------------------------------------------------------------------------
// SpMM: one wave per edge; lane moves 8 contiguous floats of the 256-wide row.
// segment-sum via global_atomic_add_f32.
// ---------------------------------------------------------------------------
__global__ __launch_bounds__(256) void spmm_kernel(const int* __restrict__ src,
                                                   const int* __restrict__ dst,
                                                   const float* __restrict__ ew,
                                                   const float* __restrict__ h,
                                                   float* __restrict__ out) {
    int gid  = blockIdx.x * 256 + threadIdx.x;
    int e    = gid >> 5;
    int lane = gid & 31;
    if (e >= N_EDGES) return;
    int   s  = src[e];
    int   d  = dst[e];
    float wt = ew[e];
    if (lane == 0 && e + 64 < N_EDGES) {
        // warm L2 for an upcoming gather row (global_prefetch_b8)
        __builtin_prefetch(&h[(size_t)src[e + 64] * NHID], 0, 1);
    }
    const float4* hs = (const float4*)(h + (size_t)s * NHID) + lane * 2;
    float4 a = hs[0];
    float4 b = hs[1];
    float* o = out + (size_t)d * NHID + lane * 8;
    unsafeAtomicAdd(o + 0, wt * a.x);
    unsafeAtomicAdd(o + 1, wt * a.y);
    unsafeAtomicAdd(o + 2, wt * a.z);
    unsafeAtomicAdd(o + 3, wt * a.w);
    unsafeAtomicAdd(o + 4, wt * b.x);
    unsafeAtomicAdd(o + 5, wt * b.y);
    unsafeAtomicAdd(o + 6, wt * b.z);
    unsafeAtomicAdd(o + 7, wt * b.w);
}

// ---------------------------------------------------------------------------
// combine: support = 0.9*hi + 0.1*h0 ; keep fp32 (in-place on hi) + bf16 copy
// ---------------------------------------------------------------------------
__global__ __launch_bounds__(256) void combine_kernel(float* __restrict__ hi,
                                                      const float* __restrict__ h0,
                                                      __bf16* __restrict__ sbf) {
    size_t i = (size_t)blockIdx.x * 256 + threadIdx.x;
    float s = 0.9f * hi[i] + 0.1f * h0[i];
    hi[i] = s;
    sbf[i] = (__bf16)s;
}

// ---------------------------------------------------------------------------
// WMMA GEMM, fc0: h = relu(x_bf16 @ w0 + b0), also h0 = h.  K = 512.
// Block: 8 waves -> 32 rows x 256 cols. Wave: 16 x 64 (4 wmma tiles).
// B stays in global (runs once; w0t is L2-resident).
// ---------------------------------------------------------------------------
__global__ __launch_bounds__(256) void gemm_fc0_kernel(const __bf16* __restrict__ A,
                                                       const __bf16* __restrict__ Bt,
                                                       const float* __restrict__ bias,
                                                       float* __restrict__ h,
                                                       float* __restrict__ h0) {
    const int K = NFEAT;
    int tid = threadIdx.x;
    int wave = tid >> 5, lane = tid & 31;
    int half = lane >> 4, l15 = lane & 15;
    int mbase = (blockIdx.x * 2 + (wave >> 2)) * 16;
    int ngrp  = (wave & 3) * 64;

    const __bf16* Arow = A + (size_t)(mbase + l15) * K;
    const __bf16* Bp[4];
#pragma unroll
    for (int t = 0; t < 4; ++t)
        Bp[t] = Bt + (size_t)(ngrp + t * 16 + l15) * K + 8 * half;

    v8f c0 = {}, c1 = {}, c2 = {}, c3 = {};
    for (int k0 = 0; k0 < K; k0 += 32) {
        v16bf a = load_a_frag(Arow, k0, half);
        v16bf b0 = load16(Bp[0] + k0);
        v16bf b1 = load16(Bp[1] + k0);
        v16bf b2 = load16(Bp[2] + k0);
        v16bf b3 = load16(Bp[3] + k0);
        c0 = __builtin_amdgcn_wmma_f32_16x16x32_bf16(false, a, false, b0, (short)0, c0, false, false);
        c1 = __builtin_amdgcn_wmma_f32_16x16x32_bf16(false, a, false, b1, (short)0, c1, false, false);
        c2 = __builtin_amdgcn_wmma_f32_16x16x32_bf16(false, a, false, b2, (short)0, c2, false, false);
        c3 = __builtin_amdgcn_wmma_f32_16x16x32_bf16(false, a, false, b3, (short)0, c3, false, false);
    }

    v8f acc[4] = {c0, c1, c2, c3};
#pragma unroll
    for (int t = 0; t < 4; ++t) {
        int n = ngrp + t * 16 + l15;
        float bv = bias[n];
#pragma unroll
        for (int r = 0; r < 8; ++r) {
            int m = mbase + r + 8 * half;
            float v = fmaxf(acc[t][r] + bv, 0.0f);
            size_t idx = (size_t)m * NHID + n;
            h[idx] = v;
            h0[idx] = v;
        }
    }
}

// ---------------------------------------------------------------------------
// WMMA GEMM, GCNII layer:
//   h = relu(theta*(S@W) + (1-theta)*S + h).  K = 256. S fp32 in `sup`.
// The full 256x256 bf16 weight (128 KB) is staged into LDS once per block
// (async global->LDS when available), so B feeds come through DScnt while A
// feeds come through LOADcnt -> independent wait chains around the WMMAs.
// ---------------------------------------------------------------------------
__global__ __launch_bounds__(256) void gemm_layer_kernel(const __bf16* __restrict__ A,
                                                         const __bf16* __restrict__ Bt,
                                                         const float* __restrict__ sup,
                                                         float* __restrict__ h,
                                                         float theta) {
    const int K = NHID;
    extern __shared__ __bf16 Bs[];                 // NHID*NHID bf16 = 128 KB

    int tid = threadIdx.x;

    // ---- stage the whole weight tile into LDS -----------------------------
#if __has_builtin(__builtin_amdgcn_global_load_async_to_lds_b128) && \
    __has_builtin(__builtin_amdgcn_s_wait_asynccnt)
#pragma unroll
    for (int j = 0; j < (NHID * NHID) / (256 * 8); ++j) {      // 32 iters
        int off = (j * 256 + tid) * 8;                          // 8 bf16 = 16 B
        __builtin_amdgcn_global_load_async_to_lds_b128(
            (__attribute__((address_space(1))) v4i_vs*)(const void*)(Bt + off),
            (__attribute__((address_space(3))) v4i_vs*)(void*)&Bs[off],
            0, 0);
    }
    __builtin_amdgcn_s_wait_asynccnt(0);
#else
#pragma unroll
    for (int j = 0; j < (NHID * NHID) / (256 * 8); ++j) {
        int off = (j * 256 + tid) * 8;
        *(float4*)&Bs[off] = *(const float4*)(Bt + off);
    }
#endif
    __syncthreads();

    // ---- wave tiling -------------------------------------------------------
    int wave = tid >> 5, lane = tid & 31;
    int half = lane >> 4, l15 = lane & 15;
    int mbase = (blockIdx.x * 2 + (wave >> 2)) * 16;
    int ngrp  = (wave & 3) * 64;

    const __bf16* Arow = A + (size_t)(mbase + l15) * K;
    const __bf16* Bp[4];
#pragma unroll
    for (int t = 0; t < 4; ++t)
        Bp[t] = Bs + (ngrp + t * 16 + l15) * K + 8 * half;

    v8f c0 = {}, c1 = {}, c2 = {}, c3 = {};
    for (int k0 = 0; k0 < K; k0 += 32) {
        v16bf a = load_a_frag(Arow, k0, half);
        v16bf b0 = load16(Bp[0] + k0);
        v16bf b1 = load16(Bp[1] + k0);
        v16bf b2 = load16(Bp[2] + k0);
        v16bf b3 = load16(Bp[3] + k0);
        c0 = __builtin_amdgcn_wmma_f32_16x16x32_bf16(false, a, false, b0, (short)0, c0, false, false);
        c1 = __builtin_amdgcn_wmma_f32_16x16x32_bf16(false, a, false, b1, (short)0, c1, false, false);
        c2 = __builtin_amdgcn_wmma_f32_16x16x32_bf16(false, a, false, b2, (short)0, c2, false, false);
        c3 = __builtin_amdgcn_wmma_f32_16x16x32_bf16(false, a, false, b3, (short)0, c3, false, false);
    }

    float omt = 1.0f - theta;
    v8f acc[4] = {c0, c1, c2, c3};
#pragma unroll
    for (int t = 0; t < 4; ++t) {
        int n = ngrp + t * 16 + l15;
#pragma unroll
        for (int r = 0; r < 8; ++r) {
            int m = mbase + r + 8 * half;
            size_t idx = (size_t)m * NHID + n;
            float v = theta * acc[t][r] + omt * sup[idx] + h[idx];
            h[idx] = fmaxf(v, 0.0f);
        }
    }
}

// ---------------------------------------------------------------------------
// classifier: out = sigmoid(h @ w1 + b1), [N,16]
// ---------------------------------------------------------------------------
__global__ __launch_bounds__(256) void classifier_kernel(const float* __restrict__ h,
                                                         const float* __restrict__ w1,
                                                         const float* __restrict__ b1,
                                                         float* __restrict__ out) {
    int i = blockIdx.x * 256 + threadIdx.x;
    if (i >= N_NODES * NCLASS) return;
    int m = i >> 4;
    int c = i & 15;
    const float* hr = h + (size_t)m * NHID;
    float acc = b1[c];
#pragma unroll 4
    for (int k = 0; k < NHID; ++k) acc += hr[k] * w1[k * NCLASS + c];
    out[i] = 1.0f / (1.0f + __expf(-acc));
}

// ---------------------------------------------------------------------------
extern "C" void kernel_launch(void* const* d_in, const int* in_sizes, int n_in,
                              void* d_out, int out_size, void* d_ws, size_t ws_size,
                              hipStream_t stream) {
    const float* x       = (const float*)d_in[0];
    const int*   e_src   = (const int*)d_in[1];
    const int*   e_dst   = (const int*)d_in[2];
    const float* e_w     = (const float*)d_in[3];
    const float* w0      = (const float*)d_in[4];
    const float* b0      = (const float*)d_in[5];
    const float* conv_w  = (const float*)d_in[6];
    const float* w1      = (const float*)d_in[7];
    const float* b1      = (const float*)d_in[8];

    const size_t NH = (size_t)N_NODES * NHID;          // 25.6M elems
    char* ws = (char*)d_ws;
    float*  h    = (float*)ws;                         // NH f32
    float*  h0   = (float*)(ws + NH * 4);              // NH f32
    float*  hi   = (float*)(ws + 2 * NH * 4);          // NH f32 (spmm accum / support)
    __bf16* xbf  = (__bf16*)(ws + 2 * NH * 4);         // aliases hi: N*512 bf16 == NH*4 bytes
    __bf16* sbf  = (__bf16*)(ws + 3 * NH * 4);         // NH bf16
    __bf16* w0t  = (__bf16*)(ws + 3 * NH * 4 + NH * 2);
    __bf16* cwt  = (__bf16*)((char*)w0t + (size_t)NFEAT * NHID * 2);

    // one-time weight/feature conversion
    cvt_f32_bf16_kernel<<<(N_NODES * NFEAT) / 256, 256, 0, stream>>>(x, xbf, N_NODES * NFEAT);
    transpose_bf16_kernel<<<(NFEAT * NHID) / 256, 256, 0, stream>>>(w0, w0t, NFEAT, NHID);
    transpose_cw_kernel<<<(NLAYERS * NHID * NHID) / 256, 256, 0, stream>>>(conv_w, cwt);

    // h = relu(x @ w0 + b0); h0 = h
    gemm_fc0_kernel<<<N_NODES / 32, 256, 0, stream>>>(xbf, w0t, b0, h, h0);

    const size_t lds_bytes = (size_t)NHID * NHID * sizeof(__bf16);   // 128 KB
    for (int i = 0; i < NLAYERS; ++i) {
        float theta = logf(0.5f / (float)(i + 1) + 1.0f);
        (void)hipMemsetAsync(hi, 0, NH * 4, stream);
        spmm_kernel<<<(N_EDGES * 32) / 256, 256, 0, stream>>>(e_src, e_dst, e_w, h, hi);
        combine_kernel<<<N_NODES, 256, 0, stream>>>(hi, h0, sbf);
        gemm_layer_kernel<<<N_NODES / 32, 256, lds_bytes, stream>>>(
            sbf, cwt + (size_t)i * NHID * NHID, hi, h, theta);
    }

    classifier_kernel<<<(N_NODES * NCLASS) / 256, 256, 0, stream>>>(h, w1, b1, (float*)d_out);
}
